// AttentionModule_57303453663460
// MI455X (gfx1250) — matrix operand order repaired and checked by reading, not compiled, not run on hardware
//
#include <hip/hip_runtime.h>

// ---------------------------------------------------------------------------
// Problem constants (reference): B=8, C=256, H=W=64 -> HW=4096, CQK=32.
// ---------------------------------------------------------------------------
#define BATCH 8
#define CCH   256
#define HWN   4096
#define CQK   32
#define OTOT  320   // 32 (q) + 32 (k) + 256 (v) stacked output channels

typedef __attribute__((ext_vector_type(16))) __bf16 v16bf;
typedef __attribute__((ext_vector_type(8)))  float  v8f;
typedef __attribute__((ext_vector_type(4)))  unsigned int v4u;
typedef __attribute__((ext_vector_type(8)))  int v8i;
typedef __attribute__((ext_vector_type(4)))  int v4i;

union FragA { v16bf v; uint4 q[2]; };   // 32B A/B matrix fragment
union FragC { v8f   v; float f[8]; };   // 16x16 f32 C/D tile

#if defined(__gfx1250__) && __has_builtin(__builtin_amdgcn_tensor_load_to_lds) && __has_builtin(__builtin_amdgcn_s_wait_tensorcnt)
#define USE_TDM 1
#else
#define USE_TDM 0
#endif

// ---------------------------------------------------------------------------
// Prep 1: concatenate {w1,w2,w3} -> bf16 Wcat[320][256], biases -> bcat[320].
// ---------------------------------------------------------------------------
__global__ __launch_bounds__(256) void prep_w_kernel(
    const float* __restrict__ w1, const float* __restrict__ b1,
    const float* __restrict__ w2, const float* __restrict__ b2,
    const float* __restrict__ w3, const float* __restrict__ b3,
    void* wcatv, float* __restrict__ bcat)
{
    __bf16* wcat = (__bf16*)wcatv;
    const int o = blockIdx.x;
    const int c = threadIdx.x;
    const float* src; const float* bsrc; int oo;
    if (o < 32)      { src = w1; bsrc = b1; oo = o; }
    else if (o < 64) { src = w2; bsrc = b2; oo = o - 32; }
    else             { src = w3; bsrc = b3; oo = o - 64; }
    wcat[o * CCH + c] = (__bf16)src[oo * CCH + c];
    if (c == 0) bcat[o] = bsrc[oo];
}

// ---------------------------------------------------------------------------
// Prep 2: transpose+convert x[b][c][i] f32 -> xT[b][i][c] bf16.
// ---------------------------------------------------------------------------
__global__ __launch_bounds__(256) void xpose_kernel(
    const float* __restrict__ x, void* xtv)
{
    __shared__ __bf16 tile[64][66];
    __bf16* xt = (__bf16*)xtv;
    const int bb = blockIdx.z;
    const int c0 = blockIdx.y * 64;
    const int i0 = blockIdx.x * 64;
    const int t  = threadIdx.x;

    const int il = t & 63, cq = t >> 6;
    #pragma unroll
    for (int rr = 0; rr < 16; ++rr) {
        const int cl = rr * 4 + cq;
        tile[cl][il] = (__bf16)x[((size_t)bb * CCH + c0 + cl) * HWN + i0 + il];
    }
    __syncthreads();
    const int cl2 = t & 63, iq = t >> 6;
    #pragma unroll
    for (int rr = 0; rr < 16; ++rr) {
        const int il2 = rr * 4 + iq;
        xt[((size_t)bb * HWN + i0 + il2) * CCH + c0 + cl2] = tile[cl2][il2];
    }
}

// ---------------------------------------------------------------------------
// Kernel 3: QKV projection, pure global->WMMA (no LDS, no barriers).
// All fragment loads are ONE per-lane base pointer + immediate offsets.
// ---------------------------------------------------------------------------
__global__ __launch_bounds__(256) void qkv_proj_kernel(
    const void* wcatv, const float* __restrict__ bcat, const void* xtv,
    void* qtv, void* ktv, void* vtv)
{
    const __bf16* __restrict__ wcat = (const __bf16*)wcatv;
    const __bf16* __restrict__ xt   = (const __bf16*)xtv;
    __bf16* __restrict__ qt = (__bf16*)qtv;
    __bf16* __restrict__ kt = (__bf16*)ktv;
    __bf16* __restrict__ vt = (__bf16*)vtv;

    const int bb   = blockIdx.y;
    const int i0   = blockIdx.x * 64;
    const int lane = threadIdx.x & 31;
    const int wv   = threadIdx.x >> 5;
    const int h    = lane >> 4;
    const int ln   = lane & 15;
    const int nt   = wv & 3;
    const int mtb  = (wv >> 2) * 10;

    FragC acc[10];
    #pragma unroll
    for (int m = 0; m < 10; ++m)
        #pragma unroll
        for (int r = 0; r < 8; ++r) acc[m].f[r] = 0.f;

    // per-lane bases (row stride 512B); everything else is immediate offsets
    const char* xbase = (const char*)(xt + ((size_t)bb * HWN + i0 + nt * 16 + ln) * CCH)
                        + 32 * h;                             // B row + K-half
    const char* wbase = (const char*)wcat + ((size_t)(mtb * 16 + ln)) * 512
                        + 32 * h;                             // A row + K-half

    #pragma unroll
    for (int step = 0; step < 8; ++step) {
        const int cb = step * 64;   // byte offset of this 32-channel chunk
        FragA fb;                   // B: col i, K bytes [32h, 32h+32)
        fb.q[0] = *(const uint4*)(xbase + cb);
        fb.q[1] = *(const uint4*)(xbase + cb + 16);
        #pragma unroll
        for (int m = 0; m < 10; ++m) {
            FragA fa;               // A: row o, K bytes [32h, 32h+32)
            fa.q[0] = *(const uint4*)(wbase + m * 8192 + cb);
            fa.q[1] = *(const uint4*)(wbase + m * 8192 + cb + 16);
            acc[m].v = __builtin_amdgcn_wmma_f32_16x16x32_bf16(
                false, fa.v, false, fb.v, (short)0, acc[m].v, false, false);
        }
    }

    // epilogue: bias + scatter bf16; destination branch is wave-uniform (mt)
    const int i = i0 + nt * 16 + ln;
    #pragma unroll
    for (int m = 0; m < 10; ++m) {
        const int mt = mtb + m;
        #pragma unroll
        for (int r = 0; r < 8; ++r) {
            const int ol = r + 8 * h;                 // C/D row within tile
            const float val = acc[m].f[r] + bcat[mt * 16 + ol];
            if (mt < 2) {
                qt[((size_t)bb * HWN + i) * CQK + mt * 16 + ol] = (__bf16)val;
            } else if (mt < 4) {
                kt[((size_t)bb * HWN + i) * CQK + (mt - 2) * 16 + ol] = (__bf16)val;
            } else {
                vt[((size_t)bb * CCH + (mt - 4) * 16 + ol) * HWN + i] = (__bf16)val;
            }
        }
    }
}

#if USE_TDM
// TDM copy of one V chunk: global vt tile [256 c][32 j] bf16 -> LDS rows of
// 64B. 2-D descriptor: dim0=j (4096, stride 4096), dim1=c. 6-arg builtin.
__device__ __forceinline__ void tdm_load_vchunk(unsigned long long gaddr,
                                                unsigned ldsaddr)
{
    v4u g0;
    g0[0] = 1u;                                        // count=1 (valid)
    g0[1] = ldsaddr;                                   // lds_addr (bytes)
    g0[2] = (unsigned)(gaddr & 0xFFFFFFFFu);           // global_addr[31:0]
    g0[3] = (unsigned)((gaddr >> 32) & 0x01FFFFFFu)    // global_addr[56:32]
          | (2u << 30);                                // type=2 (image)
    v8i g1;
    g1[0] = (int)(1u << 16);       // workgroup_mask=0, data_size=1 (2 bytes)
    g1[1] = (int)(4096u << 16);    // tensor_dim0[15:0]=4096
    g1[2] = (int)(256u << 16);     // tensor_dim0[31:16]=0 | tensor_dim1=256
    g1[3] = (int)(32u << 16);      // tensor_dim1[31:16]=0 | tile_dim0=32
    g1[4] = 256;                   // tile_dim1=256, tile_dim2=0
    g1[5] = 4096;                  // tensor_dim0_stride = 4096 elements
    g1[6] = 0;
    g1[7] = 0;
    v4i z4 = {0, 0, 0, 0};
    v8i z8 = {0, 0, 0, 0, 0, 0, 0, 0};
    __builtin_amdgcn_tensor_load_to_lds(g0, g1, z4, z4, z8, 0);
}
#endif

// ---------------------------------------------------------------------------
// Kernel 4: flash attention, TDM double-buffered V staging.
// Key-chunk loop unrolled x2 so buffer parity (and all LDS addresses) are
// compile-time constants.
// ---------------------------------------------------------------------------
__global__ __launch_bounds__(256) void attn_kernel(
    const float* __restrict__ x,
    const void* qtv, const void* ktv, const void* vtv,
    float* __restrict__ out)
{
    const __bf16* __restrict__ qt = (const __bf16*)qtv;
    const __bf16* __restrict__ kt = (const __bf16*)ktv;
    const __bf16* __restrict__ vt = (const __bf16*)vtv;

    __shared__ __align__(16) __bf16 Vlds[2 * CCH * 32];   // 2 x 16KB
    __shared__ __align__(16) __bf16 Plds[8 * 16 * 32];    // per-wave P scratch

    const int bb    = blockIdx.y;
    const int lane  = threadIdx.x & 31;
    const int wv    = threadIdx.x >> 5;
    const int h     = lane >> 4;
    const int ln    = lane & 15;
    const int qtl   = wv >> 1;
    const int chalf = wv & 1;
    const int i0    = (blockIdx.x * 4 + qtl) * 16;

    // persistent Q fragment (A: 16 queries x 32 channels)
    FragA fq;
    {
        const char* qb = (const char*)(qt + ((size_t)bb * HWN + i0 + ln) * CQK);
        fq.q[0] = *(const uint4*)(qb + 32 * h);
        fq.q[1] = *(const uint4*)(qb + 32 * h + 16);
    }

    FragC O[8];
    #pragma unroll
    for (int m = 0; m < 8; ++m)
        #pragma unroll
        for (int r = 0; r < 8; ++r) O[m].f[r] = 0.f;
    float rowm[8], rowl[8];
    #pragma unroll
    for (int r = 0; r < 8; ++r) { rowm[r] = -3.0e38f; rowl[r] = 0.f; }

    // hoisted per-lane bases
    const char* kp = (const char*)(kt + ((size_t)bb * HWN + ln) * CQK) + 32 * h;
    const char* vfrag = (const char*)Vlds + (chalf * 128 + ln) * 64 + 32 * h;
    __bf16*     myP   = Plds + wv * (16 * 32);
    const char* pld   = (const char*)(myP + ln * 32) + 32 * h;
    const __bf16* vbat = vt + (size_t)bb * CCH * HWN;

    const int NCH = HWN / 32;
#if USE_TDM
    const unsigned long long vg0 = (unsigned long long)(size_t)vbat;
    if (wv == 0)
        tdm_load_vchunk(vg0, (unsigned)(size_t)(void*)(Vlds));
#endif

    #pragma unroll 2
    for (int n = 0; n < NCH; ++n) {
        const int par = n & 1;
        const int j0  = n * 32;

#if USE_TDM
        if (wv == 0) {
            if (n + 1 < NCH) {
                tdm_load_vchunk(vg0 + (unsigned long long)(n + 1) * 64,
                                (unsigned)(size_t)(void*)(Vlds + (par ^ 1) * (CCH * 32)));
                __builtin_amdgcn_s_wait_tensorcnt(1);   // chunk n has landed
            } else {
                __builtin_amdgcn_s_wait_tensorcnt(0);
            }
        }
#else
        {   // cooperative load: 128 rows x 32 j bf16 = 512 x 16B
            __bf16* Vw = Vlds + par * (CCH * 32);
            #pragma unroll
            for (int kk = 0; kk < 2; ++kk) {
                int s = threadIdx.x + kk * 256;
                int c = s >> 2, js = s & 3;
                *(uint4*)((char*)Vw + c * 64 + js * 16) =
                    *(const uint4*)((const char*)(vbat + (size_t)c * HWN + j0) + js * 16);
            }
        }
#endif
        __syncthreads();

        // ---- S = Q^T K : two 16x16 tiles over 32 keys ----
        FragC S0, S1;
        #pragma unroll
        for (int r = 0; r < 8; ++r) { S0.f[r] = 0.f; S1.f[r] = 0.f; }
        {
            FragA fk0, fk1;
            fk0.q[0] = *(const uint4*)(kp);
            fk0.q[1] = *(const uint4*)(kp + 16);
            fk1.q[0] = *(const uint4*)(kp + 1024);
            fk1.q[1] = *(const uint4*)(kp + 1040);
            S0.v = __builtin_amdgcn_wmma_f32_16x16x32_bf16(
                false, fq.v, false, fk0.v, (short)0, S0.v, false, false);
            S1.v = __builtin_amdgcn_wmma_f32_16x16x32_bf16(
                false, fq.v, false, fk1.v, (short)0, S1.v, false, false);
        }
        kp += 2048;   // 32 keys * 64B rows

        // ---- online softmax (rows r+8h; reductions stay in 16-lane half) ----
        float corr[8];
        #pragma unroll
        for (int r = 0; r < 8; ++r) {
            float mx = fmaxf(S0.f[r], S1.f[r]);
            mx = fmaxf(mx, __shfl_xor(mx, 1, 32));
            mx = fmaxf(mx, __shfl_xor(mx, 2, 32));
            mx = fmaxf(mx, __shfl_xor(mx, 4, 32));
            mx = fmaxf(mx, __shfl_xor(mx, 8, 32));
            const float mnew = fmaxf(rowm[r], mx);
            const float c = __expf(rowm[r] - mnew);
            rowm[r] = mnew;
            const float p0 = __expf(S0.f[r] - mnew);
            const float p1 = __expf(S1.f[r] - mnew);
            S0.f[r] = p0; S1.f[r] = p1;
            float rs = p0 + p1;
            rs += __shfl_xor(rs, 1, 32);
            rs += __shfl_xor(rs, 2, 32);
            rs += __shfl_xor(rs, 4, 32);
            rs += __shfl_xor(rs, 8, 32);
            rowl[r] = rowl[r] * c + rs;
            corr[r] = c;
        }
        #pragma unroll
        for (int m = 0; m < 8; ++m)
            #pragma unroll
            for (int r = 0; r < 8; ++r) O[m].f[r] *= corr[r];

        // ---- P: C-layout -> A-layout via per-wave LDS round trip ----
        #pragma unroll
        for (int r = 0; r < 8; ++r) {
            const int mr = r + 8 * h;
            myP[mr * 32 + ln]      = (__bf16)S0.f[r];
            myP[mr * 32 + 16 + ln] = (__bf16)S1.f[r];
        }
        FragA fp;   // DS ops are in-order within a wave
        fp.q[0] = *(const uint4*)(pld);
        fp.q[1] = *(const uint4*)(pld + 16);

        // ---- O += P*V over 8 channel tiles (V-frag load rotated ahead) ----
        FragA fvc;
        fvc.q[0] = *(const uint4*)(vfrag + par * 16384);
        fvc.q[1] = *(const uint4*)(vfrag + par * 16384 + 16);
        #pragma unroll
        for (int m = 0; m < 8; ++m) {
            FragA fvn = fvc;
            if (m < 7) {
                fvn.q[0] = *(const uint4*)(vfrag + par * 16384 + (m + 1) * 1024);
                fvn.q[1] = *(const uint4*)(vfrag + par * 16384 + (m + 1) * 1024 + 16);
            }
            O[m].v = __builtin_amdgcn_wmma_f32_16x16x32_bf16(
                false, fp.v, false, fvc.v, (short)0, O[m].v, false, false);
            fvc = fvn;
        }
        __syncthreads();
    }

    // ---- epilogue: O * (1/l) + residual x ----
    float invl[8];
    #pragma unroll
    for (int r = 0; r < 8; ++r) invl[r] = 1.0f / rowl[r];
    #pragma unroll
    for (int m = 0; m < 8; ++m) {
        const int cch = (chalf * 8 + m) * 16 + ln;
        #pragma unroll
        for (int r = 0; r < 8; ++r) {
            const int i = i0 + r + 8 * h;
            const size_t idx = ((size_t)bb * CCH + cch) * HWN + i;
            out[idx] = O[m].f[r] * invl[r] + x[idx];
        }
    }
}

// ---------------------------------------------------------------------------
extern "C" void kernel_launch(void* const* d_in, const int* in_sizes, int n_in,
                              void* d_out, int out_size, void* d_ws, size_t ws_size,
                              hipStream_t stream) {
    (void)in_sizes; (void)n_in; (void)out_size; (void)ws_size;
    const float* x  = (const float*)d_in[0];
    const float* w1 = (const float*)d_in[1];
    const float* b1 = (const float*)d_in[2];
    const float* w2 = (const float*)d_in[3];
    const float* b2 = (const float*)d_in[4];
    const float* w3 = (const float*)d_in[5];
    const float* b3 = (const float*)d_in[6];
    float* out = (float*)d_out;

    char* ws = (char*)d_ws;
    const size_t qk_bytes = (size_t)BATCH * HWN * CQK * 2;        // 2 MB
    const size_t v_bytes  = (size_t)BATCH * CCH * HWN * 2;        // 16 MB
    const size_t xt_bytes = (size_t)BATCH * HWN * CCH * 2;        // 16 MB
    void*  qt   = (void*)(ws);
    void*  kt   = (void*)(ws + qk_bytes);
    void*  vt   = (void*)(ws + 2 * qk_bytes);
    void*  xt   = (void*)(ws + 2 * qk_bytes + v_bytes);
    void*  wcat = (void*)(ws + 2 * qk_bytes + v_bytes + xt_bytes);
    float* bcat = (float*)(ws + 2 * qk_bytes + v_bytes + xt_bytes
                              + (size_t)OTOT * CCH * 2);

    prep_w_kernel<<<dim3(OTOT), 256, 0, stream>>>(w1, b1, w2, b2, w3, b3, wcat, bcat);
    xpose_kernel<<<dim3(HWN / 64, CCH / 64, BATCH), 256, 0, stream>>>(x, xt);
    qkv_proj_kernel<<<dim3(HWN / 64, BATCH), 256, 0, stream>>>(wcat, bcat, xt, qt, kt, vt);
    attn_kernel<<<dim3(HWN / 64, BATCH), 256, 0, stream>>>(x, qt, kt, vt, out);
}
